// MultiScaleCosineFusion_17746804867779
// MI455X (gfx1250) — matrix operand order repaired and checked by reading, not compiled, not run on hardware
//
#include <hip/hip_runtime.h>

typedef __attribute__((ext_vector_type(16))) __bf16       v16bf;
typedef __attribute__((ext_vector_type(2)))  __bf16       v2bf;
typedef __attribute__((ext_vector_type(8)))  float        v8f;
typedef __attribute__((ext_vector_type(4)))  float        v4f;
typedef __attribute__((ext_vector_type(4)))  unsigned int v4u;
typedef __attribute__((ext_vector_type(2)))  unsigned int v2u;

union AFrag {
    v16bf bf;
    v4u   u[2];   // 32 bytes total
};

#define BDIM      64
#define DDIM      128
#define TILE_N    128
#define BF_STRIDE 136   // bf16 tile row stride (elems): 272B -> rows rotate 4 banks

// ---------------------------------------------------------------------------
// Kernel 1: normalize z_ego, split into bf16 hi/lo (row-major [64][128]),
// and compute scale = sum(softmax(w)/temps). One block of 64 threads.
// ---------------------------------------------------------------------------
__global__ void mscf_prep(const float* __restrict__ z,
                          const float* __restrict__ w,
                          unsigned short* __restrict__ a_hi,
                          unsigned short* __restrict__ a_lo,
                          float* __restrict__ scale_out)
{
    const int r = threadIdx.x;            // row 0..63
    const float* p = z + r * DDIM;

    float s = 0.f;
#pragma unroll 8
    for (int k = 0; k < DDIM; ++k) { float v = p[k]; s += v * v; }
    const float inv = 1.0f / fmaxf(sqrtf(s), 1e-8f);

#pragma unroll 8
    for (int k = 0; k < DDIM; ++k) {
        float f = p[k] * inv;
        __bf16 h = (__bf16)f;                       // RNE hi
        unsigned short hb = __builtin_bit_cast(unsigned short, h);
        float hf = __uint_as_float((unsigned)hb << 16);
        __bf16 lo = (__bf16)(f - hf);               // residual
        a_hi[r * DDIM + k] = hb;
        a_lo[r * DDIM + k] = __builtin_bit_cast(unsigned short, lo);
    }

    if (r == 0) {
        float w0 = w[0], w1 = w[1], w2 = w[2];
        float m  = fmaxf(w0, fmaxf(w1, w2));
        float e0 = __expf(w0 - m), e1 = __expf(w1 - m), e2 = __expf(w2 - m);
        float is = 1.0f / (e0 + e1 + e2);
        scale_out[0] = is * (e0 / 0.07f + e1 / 0.1f + e2 / 0.2f);
    }
}

// ---------------------------------------------------------------------------
// Kernel 2: per block, one 128-column tile of exo_bank.
//   phase 1 (fused): global->register loads, sum-of-squares, bf16 hi/lo split
//            of RAW exo values straight into shared LDS tiles
//   phase 2: invn[row] = 1/max(||x||,eps)
//   phase 3: WMMA split-product GEMM, out = c * scale * invn[col]
// ---------------------------------------------------------------------------
__global__ void __launch_bounds__(256)
mscf_gemm(const float* __restrict__ exo,
          const unsigned short* __restrict__ a_hi,
          const unsigned short* __restrict__ a_lo,
          const float* __restrict__ scale_p,
          float* __restrict__ out,
          int Nn)
{
    __shared__ unsigned short bhi_t[TILE_N * BF_STRIDE];      // 34816 B
    __shared__ unsigned short blo_t[TILE_N * BF_STRIDE];      // 34816 B
    __shared__ float          partial[TILE_N][4];
    __shared__ float          invn[TILE_N];

    const int n0 = blockIdx.x * TILE_N;

    // ---- phase 1 (fused): load raw exo, sumsq, split to bf16 hi/lo tiles ----
    // 4 threads per row, 32 elems each; two 64-row groups (128 rows total).
#pragma unroll
    for (int g = 0; g < 2; ++g) {
        const int row = (g << 6) + (threadIdx.x >> 2);
        const int q   = threadIdx.x & 3;
        unsigned gn = (unsigned)(n0 + row);
        if (gn > (unsigned)(Nn - 1)) gn = (unsigned)(Nn - 1);  // tail clamp
        // 32-bit offset math: max exo byte offset = 512MB < 2^31
        const float*    gp = exo + gn * (unsigned)DDIM + (unsigned)(q << 5);
        unsigned short* ph = &bhi_t[row * BF_STRIDE + (q << 5)];
        unsigned short* pl = &blo_t[row * BF_STRIDE + (q << 5)];
        float s = 0.f;
#pragma unroll
        for (int c = 0; c < 32; c += 4) {
            v4f x = *(const v4f*)(gp + c);
            float f0 = x[0], f1 = x[1], f2 = x[2], f3 = x[3];
            s += f0 * f0; s += f1 * f1; s += f2 * f2; s += f3 * f3;
            v2bf h01 = { (__bf16)f0, (__bf16)f1 };
            v2bf h23 = { (__bf16)f2, (__bf16)f3 };
            unsigned hw0 = __builtin_bit_cast(unsigned, h01);
            unsigned hw1 = __builtin_bit_cast(unsigned, h23);
            float hf0 = __uint_as_float(hw0 << 16);
            float hf1 = __uint_as_float(hw0 & 0xFFFF0000u);
            float hf2 = __uint_as_float(hw1 << 16);
            float hf3 = __uint_as_float(hw1 & 0xFFFF0000u);
            v2bf l01 = { (__bf16)(f0 - hf0), (__bf16)(f1 - hf1) };
            v2bf l23 = { (__bf16)(f2 - hf2), (__bf16)(f3 - hf3) };
            v2u hv = { hw0, hw1 };
            v2u lv = { __builtin_bit_cast(unsigned, l01),
                       __builtin_bit_cast(unsigned, l23) };
            *(v2u*)(ph + c) = hv;          // ds_store_b64
            *(v2u*)(pl + c) = lv;          // ds_store_b64
        }
        partial[row][q] = s;
    }
    __syncthreads();

    // ---- phase 2: per-row inverse norms ----
    if (threadIdx.x < TILE_N) {
        float s = partial[threadIdx.x][0] + partial[threadIdx.x][1]
                + partial[threadIdx.x][2] + partial[threadIdx.x][3];
        invn[threadIdx.x] = 1.0f / fmaxf(sqrtf(s), 1e-8f);
    }
    __syncthreads();

    // ---- phase 3: WMMA ----
    const int lane   = threadIdx.x & 31;
    const int wave   = threadIdx.x >> 5;
    const int mtile  = wave & 3;          // 4 tiles cover B=64 rows
    const int ntbase = wave >> 2;         // ntiles {ntbase, +2, +4, +6} of 8
    const int l15    = lane & 15;
    const int lhalf  = lane >> 4;

    // A fragment: 16-bit A 16x32 layout — lane<16: K{0..7,16..23}, lane>=16: K{8..15,24..31}
    const int khalfA = lhalf << 3;
    const int mrow   = (mtile << 4) + l15;

    AFrag ahi[4], alo[4];
#pragma unroll
    for (int kk = 0; kk < 4; ++kk) {
        unsigned off0 = (unsigned)(mrow * DDIM + (kk << 5) + khalfA);
        ahi[kk].u[0] = *(const v4u*)(a_hi + off0);
        ahi[kk].u[1] = *(const v4u*)(a_hi + off0 + 16);
        alo[kk].u[0] = *(const v4u*)(a_lo + off0);
        alo[kk].u[1] = *(const v4u*)(a_lo + off0 + 16);
    }

    const float scale = *scale_p;

#pragma unroll
    for (int j = 0; j < 4; ++j) {
        const int ntile = ntbase + (j << 1);
        const int brow  = (ntile << 4) + l15;   // local exo row = this lane's B column

        v8f c = {0.f, 0.f, 0.f, 0.f, 0.f, 0.f, 0.f, 0.f};
#pragma unroll
        for (int kk = 0; kk < 4; ++kk) {
            // dense 16-bit B 32x16 layout (matches A): lanes 0-15 hold K{0..7,16..23},
            // lanes 16-31 hold K{8..15,24..31}
            const int boff = brow * BF_STRIDE + (kk << 5) + (lhalf << 3);
            AFrag bhi, blo;
            bhi.u[0] = *(const v4u*)(bhi_t + boff);
            bhi.u[1] = *(const v4u*)(bhi_t + boff + 16);
            blo.u[0] = *(const v4u*)(blo_t + boff);
            blo.u[1] = *(const v4u*)(blo_t + boff + 16);
            // split product: hi*hi + hi*lo + lo*hi, fp32 accumulate
            c = __builtin_amdgcn_wmma_f32_16x16x32_bf16(
                    false, ahi[kk].bf, false, bhi.bf, (short)0, c, false, false);
            c = __builtin_amdgcn_wmma_f32_16x16x32_bf16(
                    false, ahi[kk].bf, false, blo.bf, (short)0, c, false, false);
            c = __builtin_amdgcn_wmma_f32_16x16x32_bf16(
                    false, alo[kk].bf, false, bhi.bf, (short)0, c, false, false);
        }

        // C/D layout: VGPR r -> M = r (lanes 0-15) or r+8 (lanes 16-31), N = lane&15
        const int ncol = n0 + (ntile << 4) + l15;
        if (ncol < Nn) {
            const float sc = scale * invn[brow];    // fold 1/||x|| in at output
            const int mbase = (mtile << 4) + (lhalf << 3);
            // 32-bit offset math: max out byte offset = 256MB < 2^31
            unsigned off = (unsigned)mbase * (unsigned)Nn + (unsigned)ncol;
#pragma unroll
            for (int r = 0; r < 8; ++r) {
                out[off] = c[r] * sc;
                off += (unsigned)Nn;
            }
        }
    }
}

// ---------------------------------------------------------------------------
extern "C" void kernel_launch(void* const* d_in, const int* in_sizes, int n_in,
                              void* d_out, int out_size, void* d_ws, size_t ws_size,
                              hipStream_t stream)
{
    const float* z   = (const float*)d_in[0];   // [64,128] f32
    const float* exo = (const float*)d_in[1];   // [N,128]  f32
    const float* w   = (const float*)d_in[2];   // [3]      f32
    float* out = (float*)d_out;                 // [64,N]   f32

    const int Nn = in_sizes[1] / DDIM;

    // workspace layout: [scale f32 (pad 64B)] [a_hi bf16 64*128] [a_lo bf16 64*128]
    char* ws = (char*)d_ws;
    float*          scale_p = (float*)ws;
    unsigned short* a_hi    = (unsigned short*)(ws + 64);
    unsigned short* a_lo    = (unsigned short*)(ws + 64 + BDIM * DDIM * 2);

    mscf_prep<<<1, BDIM, 0, stream>>>(z, w, a_hi, a_lo, scale_p);

    const int nblocks = (Nn + TILE_N - 1) / TILE_N;
    mscf_gemm<<<nblocks, 256, 0, stream>>>(exo, a_hi, a_lo, scale_p, out, Nn);
}
